// Decoder_35794257445248
// MI455X (gfx1250) — matrix operand order repaired and checked by reading, not compiled, not run on hardware
//
#include <hip/hip_runtime.h>

// ---------------- problem constants ----------------
#define BB   16
#define NN   325
#define DD   64
#define HH   4
#define DHH  16
#define SEQ  12
#define NPAD 352            // 325 padded up to multiple of 32
#define BND  (BB*NN*DD)     // 332800
#define ITILES 21           // ceil(325/16)

typedef __attribute__((ext_vector_type(16))) _Float16 v16h;
typedef __attribute__((ext_vector_type(8)))  _Float16 v8h;
typedef __attribute__((ext_vector_type(8)))  float    v8f;
typedef __attribute__((ext_vector_type(2)))  float    v2f;
typedef __attribute__((ext_vector_type(4)))  unsigned int v4u;
typedef __attribute__((ext_vector_type(8)))  int      v8i;
typedef __attribute__((ext_vector_type(4)))  int      v4i;

#if __has_builtin(__builtin_amdgcn_tensor_load_to_lds)
#define USE_TDM 1
#else
#define USE_TDM 0
#endif

// ---------------- elementwise kernels ----------------
__global__ void bias_kernel(const float* __restrict__ g, float* __restrict__ bias) {
  int idx = blockIdx.x * blockDim.x + threadIdx.x;
  if (idx >= NN * NN) return;
  int i = idx / NN, j = idx - i * NN;
  bool m = (g[idx] > 0.9f) || (i == j);
  bias[idx] = m ? 0.0f : -1.0e9f;
}

__global__ void copy_kernel(const float* __restrict__ in, float* __restrict__ out, int n) {
  int idx = blockIdx.x * blockDim.x + threadIdx.x;
  if (idx < n) out[idx] = in[idx];
}

__global__ void zero_h16_kernel(_Float16* __restrict__ h16, int n) {
  int idx = blockIdx.x * blockDim.x + threadIdx.x;
  if (idx < n) h16[idx] = (_Float16)0.0f;
}

__global__ void elu_kernel(float* __restrict__ x, int n) {
  int idx = blockIdx.x * blockDim.x + threadIdx.x;
  if (idx < n) {
    float v = x[idx];
    x[idx] = v > 0.0f ? v : (__expf(v) - 1.0f);
  }
}

__global__ void axpy_kernel(const float* __restrict__ y, const float* __restrict__ k,
                            float c, float* __restrict__ out, int n) {
  int idx = blockIdx.x * blockDim.x + threadIdx.x;
  if (idx < n) out[idx] = y[idx] + c * k[idx];
}

__global__ void rk4_combine_kernel(const float* __restrict__ y,
                                   const float* __restrict__ k1, const float* __restrict__ k2,
                                   const float* __restrict__ k3, const float* __restrict__ k4,
                                   float c, float* __restrict__ y1, int n) {
  int idx = blockIdx.x * blockDim.x + threadIdx.x;
  if (idx < n)
    y1[idx] = y[idx] + c * (k1[idx] + 2.0f * k2[idx] + 2.0f * k3[idx] + k4[idx]);
}

// e_src[b,h,n] = sum_d h[b,n,h*16+d] * a_src[h,d]   (and e_dst likewise)
__global__ void edge_proj_kernel(const float* __restrict__ h,
                                 const float* __restrict__ asrc, const float* __restrict__ adst,
                                 float* __restrict__ esrc, float* __restrict__ edst) {
  int idx = blockIdx.x * blockDim.x + threadIdx.x;
  if (idx >= BB * NN) return;
  int b = idx / NN, n = idx - b * NN;
  const float* hr = h + (size_t)idx * DD;
  for (int hd = 0; hd < HH; ++hd) {
    float s = 0.0f, d2 = 0.0f;
    for (int d = 0; d < DHH; ++d) {
      float v = hr[hd * DHH + d];
      s  += v * asrc[hd * DHH + d];
      d2 += v * adst[hd * DHH + d];
    }
    esrc[((size_t)b * HH + hd) * NN + n] = s;
    edst[((size_t)b * HH + hd) * NN + n] = d2;
  }
}

// ------- h = x @ W (fp32 WMMA 16x16x4, A reused across 4 col tiles) -------
// h16 layout transposed for attention: h16[((b*H + head)*16 + d) * NPAD + j]
__global__ __launch_bounds__(32)
void gemm_xw_kernel(const float* __restrict__ x, const float* __restrict__ W,
                    float* __restrict__ h, _Float16* __restrict__ h16) {
  int m0 = blockIdx.x * 16;
  int lane = threadIdx.x & 31;
  int half = lane >> 4, r = lane & 15;
  v8f acc[4] = {{}, {}, {}, {}};
  for (int kc = 0; kc < 16; ++kc) {         // K = 64 in chunks of 4
    int ka = kc * 4 + half * 2;
    v2f a;
    a[0] = x[(size_t)(m0 + r) * DD + ka];
    a[1] = x[(size_t)(m0 + r) * DD + ka + 1];
    for (int ct = 0; ct < 4; ++ct) {
      v2f bfr;
      bfr[0] = W[(size_t)ka * DD + ct * 16 + r];
      bfr[1] = W[(size_t)(ka + 1) * DD + ct * 16 + r];
      acc[ct] = __builtin_amdgcn_wmma_f32_16x16x4_f32(false, a, false, bfr, (short)0,
                                                      acc[ct], false, false);
    }
  }
  for (int ct = 0; ct < 4; ++ct) {
    for (int v = 0; v < 8; ++v) {
      int row = m0 + v + half * 8;          // global row in [0, B*N)
      float val = acc[ct][v];
      h[(size_t)row * DD + ct * 16 + r] = val;
      int b = row / NN, nn = row - b * NN;  // head == ct, d == r
      h16[(((size_t)b * HH + ct) * DHH + r) * NPAD + nn] = (_Float16)val;
    }
  }
}

// ---------------- fused masked-softmax + alpha@h (f16 WMMA 16x16x32) ----------------
__global__ __launch_bounds__(128)
void attn_kernel(const _Float16* __restrict__ h16, const float* __restrict__ esrc,
                 const float* __restrict__ edst, const float* __restrict__ bias,
                 float* __restrict__ out) {
  __shared__ float    S[16 * NPAD];
  __shared__ __align__(16) _Float16 A16[16 * NPAD];
  __shared__ float    Esr[16];
  __shared__ float    Eds[NPAD];
  __shared__ float    partial[4][32][8];
#if USE_TDM
  __shared__ __align__(16) _Float16 Hs[DHH * NPAD];   // staged h tile (d-major, j-contig)
#endif

  int bid  = blockIdx.x;
  int it   = bid % ITILES;
  int rem  = bid / ITILES;
  int head = rem % HH;
  int b    = rem / HH;
  int i0   = it * 16;
  int t    = threadIdx.x;
  int lane = t & 31, wave = t >> 5;
  int half = lane >> 4, r = lane & 15;

  const _Float16* hb = h16 + ((size_t)b * HH + head) * DHH * NPAD;

#if USE_TDM
  // Stage the 16 x 352 f16 tile into LDS with the Tensor Data Mover (one op, wave 0).
  // D#: 2D tensor, data_size=8B, one row of 1408 elements (11264 bytes, contiguous).
  if (wave == 0) {
    unsigned long long ga = (unsigned long long)(size_t)hb;
    unsigned ldsa = (unsigned)(size_t)(void*)&Hs[0];
    v4u g0 = { 1u,                                    // count=1, user descriptor
               ldsa,                                  // lds_addr
               (unsigned)(ga & 0xffffffffu),          // global_addr[31:0]
               (unsigned)((ga >> 32) & 0x01ffffffu) | 0x80000000u }; // addr[56:32] | type=2
    v8i g1 = { 0x00030000,   // data_size = 3 (8 bytes), no multicast/pad/iterate
               0x05800000,   // tensor_dim0 low16 = 1408 (<<16)
               0x00010000,   // tensor_dim1 = 1 (<<16)
               0x05800000,   // tile_dim0 = 1408 (<<16)
               0x00000001,   // tile_dim1 = 1
               0x00000580,   // tensor_dim0_stride = 1408
               0x05800000,   // tensor_dim1_stride low16 = 1408 (<<16)
               0 };
    v4i g2 = { 0, 0, 0, 0 };
    v4i g3 = { 0, 0, 0, 0 };
#if __clang_major__ >= 23
    v8i g4 = { 0, 0, 0, 0, 0, 0, 0, 0 };
    __builtin_amdgcn_tensor_load_to_lds(g0, g1, g2, g3, g4, 0);
#else
    __builtin_amdgcn_tensor_load_to_lds(g0, g1, g2, g3, 0);
#endif
  }
#endif

  const float* es = esrc + ((size_t)b * HH + head) * NN;
  const float* ed = edst + ((size_t)b * HH + head) * NN;
  if (t < 16) Esr[t] = (i0 + t < NN) ? es[i0 + t] : 0.0f;
  for (int j = t; j < NPAD; j += 128) Eds[j] = (j < NN) ? ed[j] : 0.0f;
  __syncthreads();

  // scores
  for (int idx = t; idx < 16 * NPAD; idx += 128) {
    int i = idx / NPAD, j = idx - i * NPAD;
    int ni = i0 + i;
    float s;
    if (ni < NN && j < NN) {
      float e = Esr[i] + Eds[j];
      e = e > 0.0f ? e : 0.2f * e;          // leaky relu
      s = e + bias[(size_t)ni * NN + j];    // 0 or -1e9 mask
    } else {
      s = -1.0e9f;
    }
    S[idx] = s;
  }
  __syncthreads();

  // row softmax: wave w handles rows w, w+4, w+8, w+12
  for (int rr = wave; rr < 16; rr += 4) {
    float mx = -3.0e38f;
    for (int j = lane; j < NPAD; j += 32) mx = fmaxf(mx, S[rr * NPAD + j]);
    for (int m = 16; m >= 1; m >>= 1) mx = fmaxf(mx, __shfl_xor(mx, m));
    float sum = 0.0f;
    for (int j = lane; j < NPAD; j += 32) {
      float e = __expf(S[rr * NPAD + j] - mx);
      S[rr * NPAD + j] = e;
      sum += e;
    }
    for (int m = 16; m >= 1; m >>= 1) sum += __shfl_xor(sum, m);
    float inv = 1.0f / sum;
    for (int j = lane; j < NPAD; j += 32)
      A16[rr * NPAD + j] = (_Float16)(S[rr * NPAD + j] * inv);
  }
#if USE_TDM
  if (wave == 0) __builtin_amdgcn_s_wait_tensorcnt(0);  // tile resident before barrier
#endif
  __syncthreads();

  // alpha(16 x 352) @ h(352 x 16), K chunks split across waves
  v8f acc = {};
  for (int kc = wave; kc < NPAD / 32; kc += 4) {
    // A fragment: two aligned 16B LDS reads per lane
    const v8h* ap = (const v8h*)(A16 + (size_t)r * NPAD + half * 8 + kc * 32);
    v8h alo = ap[0];
    v8h ahi = ap[2];                        // +16 halves
    // B fragment: two 8-half runs at +0 and +16 (LDS-staged via TDM, or global)
#if USE_TDM
    const v8h* bp = (const v8h*)(Hs + (size_t)r * NPAD + kc * 32 + half * 8);
#else
    const v8h* bp = (const v8h*)(hb + (size_t)r * NPAD + kc * 32 + half * 8);
#endif
    v8h blo = bp[0];
    v8h bhi = bp[2];
    v16h af, bf;
    for (int i = 0; i < 8; ++i) {
      af[i] = alo[i]; af[i + 8] = ahi[i];
      bf[i] = blo[i]; bf[i + 8] = bhi[i];
    }
    acc = __builtin_amdgcn_wmma_f32_16x16x32_f16(false, af, false, bf, (short)0, acc,
                                                 false, false);
  }
  for (int v = 0; v < 8; ++v) partial[wave][lane][v] = acc[v];
  __syncthreads();

  if (wave == 0) {
    for (int v = 0; v < 8; ++v) {
      float c = partial[0][lane][v] + partial[1][lane][v] +
                partial[2][lane][v] + partial[3][lane][v];
      int ni = i0 + v + half * 8;
      if (ni < NN) out[((size_t)b * NN + ni) * DD + head * DHH + r] = c;
    }
  }
}

// ---------------- head: tanh(traj@W1 + b1) @ W2 + b2  (fp32 WMMA, fused) ----------------
__global__ __launch_bounds__(32)
void head_kernel(const float* __restrict__ traj, const float* __restrict__ W1,
                 const float* __restrict__ b1, const float* __restrict__ W2,
                 const float* __restrict__ b2, float* __restrict__ outp) {
  int m0 = blockIdx.x * 16;
  int lane = threadIdx.x & 31;
  int half = lane >> 4, r = lane & 15;
  v8f acc[4] = {{}, {}, {}, {}};
  for (int kc = 0; kc < 16; ++kc) {
    int ka = kc * 4 + half * 2;
    v2f a;
    a[0] = traj[(size_t)(m0 + r) * DD + ka];
    a[1] = traj[(size_t)(m0 + r) * DD + ka + 1];
    for (int ct = 0; ct < 4; ++ct) {
      v2f bfr;
      bfr[0] = W1[(size_t)ka * DD + ct * 16 + r];
      bfr[1] = W1[(size_t)(ka + 1) * DD + ct * 16 + r];
      acc[ct] = __builtin_amdgcn_wmma_f32_16x16x4_f32(false, a, false, bfr, (short)0,
                                                      acc[ct], false, false);
    }
  }
  float s[8] = {0, 0, 0, 0, 0, 0, 0, 0};
  for (int ct = 0; ct < 4; ++ct) {
    float bcol = b1[ct * 16 + r];
    float w2c  = W2[ct * 16 + r];
    for (int v = 0; v < 8; ++v)
      s[v] += tanhf(acc[ct][v] + bcol) * w2c;
  }
  float b2v = b2[0];
  for (int v = 0; v < 8; ++v) {
    float sv = s[v];
    sv += __shfl_xor(sv, 1);
    sv += __shfl_xor(sv, 2);
    sv += __shfl_xor(sv, 4);
    sv += __shfl_xor(sv, 8);
    if (r == 0) {
      int row = m0 + half * 8 + v;          // row in (s, b, n) order
      int sstep = row / (BB * NN);
      int rem2  = row - sstep * (BB * NN);
      int bb = rem2 / NN;
      int n  = rem2 - bb * NN;
      outp[((size_t)bb * (SEQ + 1) + sstep) * NN + n] = sv + b2v;
    }
  }
}

// ---------------- host orchestration ----------------
namespace {

struct Ctx {
  const float *Wg, *a_src, *a_dst, *bias;
  float *hbuf, *esrc, *edst;
  _Float16* h16;
  float* g0;
  hipStream_t stream;
};

inline void run_gat(const Ctx& c, const float* x, int layer, float* out) {
  const float* W  = c.Wg    + (size_t)layer * DD * DD;
  const float* as = c.a_src + (size_t)layer * HH * DHH;
  const float* ad = c.a_dst + (size_t)layer * HH * DHH;
  gemm_xw_kernel<<<BB * NN / 16, 32, 0, c.stream>>>(x, W, c.hbuf, c.h16);
  edge_proj_kernel<<<(BB * NN + 255) / 256, 256, 0, c.stream>>>(c.hbuf, as, ad, c.esrc, c.edst);
  attn_kernel<<<BB * HH * ITILES, 128, 0, c.stream>>>(c.h16, c.esrc, c.edst, c.bias, out);
}

inline void run_ode(const Ctx& c, const float* y, float* out) {
  run_gat(c, y, 0, c.g0);
  elu_kernel<<<BND / 256, 256, 0, c.stream>>>(c.g0, BND);
  run_gat(c, c.g0, 1, out);
}

}  // namespace

extern "C" void kernel_launch(void* const* d_in, const int* in_sizes, int n_in,
                              void* d_out, int out_size, void* d_ws, size_t ws_size,
                              hipStream_t stream) {
  (void)in_sizes; (void)n_in; (void)out_size; (void)ws_size;
  const float* y0    = (const float*)d_in[0];
  const float* graph = (const float*)d_in[1];
  const float* Wg    = (const float*)d_in[2];
  const float* a_src = (const float*)d_in[3];
  const float* a_dst = (const float*)d_in[4];
  const float* W1    = (const float*)d_in[5];
  const float* b1    = (const float*)d_in[6];
  const float* W2    = (const float*)d_in[7];
  const float* b2    = (const float*)d_in[8];

  // workspace carving (256B aligned)
  size_t off = 0;
  auto alloc = [&](size_t bytes) -> void* {
    off = (off + 255) & ~(size_t)255;
    void* p = (char*)d_ws + off;
    off += bytes;
    return p;
  };
  float*    bias = (float*)alloc((size_t)NN * NN * 4);
  float*    traj = (float*)alloc((size_t)(SEQ + 1) * BND * 4);
  float*    ytmp = (float*)alloc((size_t)BND * 4);
  float*    k1   = (float*)alloc((size_t)BND * 4);
  float*    k2   = (float*)alloc((size_t)BND * 4);
  float*    k3   = (float*)alloc((size_t)BND * 4);
  float*    k4   = (float*)alloc((size_t)BND * 4);
  float*    g0   = (float*)alloc((size_t)BND * 4);
  float*    hbuf = (float*)alloc((size_t)BND * 4);
  _Float16* h16  = (_Float16*)alloc((size_t)BB * HH * DHH * NPAD * 2);
  float*    esrc = (float*)alloc((size_t)BB * HH * NN * 4);
  float*    edst = (float*)alloc((size_t)BB * HH * NN * 4);

  Ctx c{Wg, a_src, a_dst, bias, hbuf, esrc, edst, h16, g0, stream};

  bias_kernel<<<(NN * NN + 255) / 256, 256, 0, stream>>>(graph, bias);
  zero_h16_kernel<<<(BB * HH * DHH * NPAD + 255) / 256, 256, 0, stream>>>(
      h16, BB * HH * DHH * NPAD);
  copy_kernel<<<BND / 256, 256, 0, stream>>>(y0, traj, BND);

  for (int s = 0; s < SEQ; ++s) {
    float dt = (float)(s + 1) * 0.1f - (float)s * 0.1f;  // match T=linspace*SCALE diffs
    float* y  = traj + (size_t)s * BND;
    float* y1 = traj + (size_t)(s + 1) * BND;
    run_ode(c, y, k1);
    axpy_kernel<<<BND / 256, 256, 0, stream>>>(y, k1, 0.5f * dt, ytmp, BND);
    run_ode(c, ytmp, k2);
    axpy_kernel<<<BND / 256, 256, 0, stream>>>(y, k2, 0.5f * dt, ytmp, BND);
    run_ode(c, ytmp, k3);
    axpy_kernel<<<BND / 256, 256, 0, stream>>>(y, k3, dt, ytmp, BND);
    run_ode(c, ytmp, k4);
    rk4_combine_kernel<<<BND / 256, 256, 0, stream>>>(y, k1, k2, k3, k4, dt / 6.0f, y1, BND);
  }

  head_kernel<<<(SEQ + 1) * BB * NN / 16, 32, 0, stream>>>(traj, W1, b1, W2, b2,
                                                           (float*)d_out);
}
